// PointNetSegmenter_16844861734918
// MI455X (gfx1250) — matrix operand order repaired and checked by reading, not compiled, not run on hardware
//
#include <hip/hip_runtime.h>
#include <limits>

// ---------------------------------------------------------------------------
// PointNet-style GNN segmenter for MI455X (gfx1250), wave32 + WMMA bf16.
//
// Per layer (fused, per 16-edge tile, one wave per tile):
//   gather edge feats -> LDS (bf16) -> WMMA x W1 (+b1) -> relu -> LDS ->
//   WMMA x W2 (+b2) -> relu -> atomicMax scatter into zero-init agg.
// relu(segment_max with -inf->0) == max(0, messages) with 0-init accumulator,
// and for non-negative floats uint atomicMax == float max.
//
// Layer kernels are templated on <D_IN, KPAD, NKC1> so all fragment arrays
// are constant-indexed and live entirely in VGPRs (no scratch spills).
// Full 16-edge tiles take a straight-line path (no per-lane exec guards).
// Relu is a single v_med3_f32 (clamp to [0, +inf]).
// ---------------------------------------------------------------------------

typedef __attribute__((ext_vector_type(16))) __bf16 v16bf;
typedef __attribute__((ext_vector_type(8)))  __bf16 v8bf;
typedef __attribute__((ext_vector_type(8)))  float  v8f;

#define HID      64
#define MAX_KPAD 96
#define WPB      8            // waves per block (256 threads)

static __device__ __forceinline__ float relu1(float v) {
    // v_med3_f32 v, 0, +inf : one-instruction relu, no canonicalize.
    return __builtin_amdgcn_fmed3f(v, 0.0f, __builtin_inff());
}

static __device__ __forceinline__ unsigned short f2bf_bits(float f) {
    unsigned int u = __float_as_uint(f);
    unsigned int r = u + 0x7FFFu + ((u >> 16) & 1u);   // round-to-nearest-even
    return (unsigned short)(r >> 16);
}

// ---------------------------------------------------------------------------
// Pack a row-major f32 weight [Kact x 64] into per-lane WMMA B fragments.
// Fragment block (kc,nt): 32 lanes x 8 dwords, lane holds column n = nt*16 +
// (lane&15), K = kc*32 + (lane>>4)*16 + {2j, 2j+1} packed bf16x2 per dword.
// ---------------------------------------------------------------------------
__global__ void pack_b_kernel(const float* __restrict__ w,
                              unsigned int* __restrict__ out,
                              int Kact, int nkc) {
    int t = blockIdx.x * blockDim.x + threadIdx.x;   // one thread per lane-slot
    int total = nkc * 4 * 32;
    if (t >= total) return;
    int lane = t & 31;
    int nt   = (t >> 5) & 3;
    int kc   = t >> 7;
    int n     = nt * 16 + (lane & 15);
    int kbase = kc * 32 + ((lane >> 4) * 16);
#pragma unroll
    for (int j = 0; j < 8; ++j) {
        int k0 = kbase + 2 * j;
        int k1 = k0 + 1;
        float f0 = (k0 < Kact) ? w[(size_t)k0 * HID + n] : 0.0f;
        float f1 = (k1 < Kact) ? w[(size_t)k1 * HID + n] : 0.0f;
        unsigned int u = (unsigned int)f2bf_bits(f0) |
                         ((unsigned int)f2bf_bits(f1) << 16);
        out[(size_t)t * 8 + j] = u;
    }
}

__global__ void zero_kernel(float4* __restrict__ p, size_t n4) {
    size_t i = (size_t)blockIdx.x * blockDim.x + threadIdx.x;
    size_t stride = (size_t)gridDim.x * blockDim.x;
    const float4 z = {0.0f, 0.0f, 0.0f, 0.0f};
    for (; i < n4; i += stride) p[i] = z;
}

// ---------------------------------------------------------------------------
// Fused edge-MLP + segment-max layer. One wave processes one 16-edge tile.
// ---------------------------------------------------------------------------
template <int D_IN, int KPAD, int NKC1>
__global__ __launch_bounds__(256) void pointnet_layer_kernel(
    const float* __restrict__ h_in,
    const float* __restrict__ pos,
    const long long* __restrict__ src_idx,
    const long long* __restrict__ dst_idx,
    const unsigned int* __restrict__ w1p, const float* __restrict__ b1,
    const unsigned int* __restrict__ w2p, const float* __restrict__ b2,
    float* __restrict__ agg,                 // [N_NODES, 64], pre-zeroed
    int n_edges)
{
    __shared__ __bf16 lds[WPB * (16 * MAX_KPAD + 16 * HID)];

    const int lane = threadIdx.x & 31;
    const int wave = threadIdx.x >> 5;
    __bf16* tileA = &lds[wave * (16 * MAX_KPAD + 16 * HID)];
    __bf16* tileY = tileA + 16 * MAX_KPAD;

    const int n0   = lane & 15;      // column within 16-wide N tile / A row
    const int half = lane >> 4;      // which half-wave (K split)

    // Preload all B fragments into VGPRs (uniform across the edge loop).
    v16bf w1f[NKC1][4];
#pragma unroll
    for (int kc = 0; kc < NKC1; ++kc)
#pragma unroll
        for (int nt = 0; nt < 4; ++nt)
            w1f[kc][nt] = *reinterpret_cast<const v16bf*>(
                w1p + ((size_t)((kc * 4 + nt) * 32 + lane)) * 8);
    v16bf w2f[2][4];
#pragma unroll
    for (int kc = 0; kc < 2; ++kc)
#pragma unroll
        for (int nt = 0; nt < 4; ++nt)
            w2f[kc][nt] = *reinterpret_cast<const v16bf*>(
                w2p + ((size_t)((kc * 4 + nt) * 32 + lane)) * 8);

    float b1n[4], b2n[4];
#pragma unroll
    for (int nt = 0; nt < 4; ++nt) {
        b1n[nt] = b1[nt * 16 + n0];
        b2n[nt] = b2[nt * 16 + n0];
    }

    const int n_tiles = (n_edges + 15) >> 4;
    const int wave_id = blockIdx.x * WPB + wave;
    const int n_waves = gridDim.x * WPB;

    for (int tile = wave_id; tile < n_tiles; tile += n_waves) {
        const int ebase = tile * 16;
        const bool full = (ebase + 16 <= n_edges);   // uniform across wave

        // ---- Stage A tile into LDS: row = n0; this half-wave writes the
        //      contiguous range k in [half*KPAD/2, (half+1)*KPAD/2).
        const int k0 = half * (KPAD / 2);
        if (full) {
            const int e = ebase + n0;
            const long long s = src_idx[e];
            const long long d = dst_idx[e];
            const float* hrow = h_in + (size_t)s * D_IN;
#pragma unroll
            for (int kk = 0; kk < KPAD / 2; ++kk) {
                const int k = k0 + kk;
                float v = 0.0f;
                if (k < D_IN)           v = hrow[k];
                else if (k < D_IN + 3)  v = pos[(size_t)s * 3 + (k - D_IN)] -
                                            pos[(size_t)d * 3 + (k - D_IN)];
                tileA[n0 * KPAD + k] = (__bf16)v;
            }
        } else {
            const int e = ebase + n0;
            long long s = 0, d = 0;
            const bool valid = (e < n_edges);
            if (valid) { s = src_idx[e]; d = dst_idx[e]; }
            const float* hrow = h_in + (size_t)s * D_IN;
#pragma unroll
            for (int kk = 0; kk < KPAD / 2; ++kk) {
                const int k = k0 + kk;
                float v = 0.0f;
                if (valid) {
                    if (k < D_IN)           v = hrow[k];
                    else if (k < D_IN + 3)  v = pos[(size_t)s * 3 + (k - D_IN)] -
                                                pos[(size_t)d * 3 + (k - D_IN)];
                }
                tileA[n0 * KPAD + k] = (__bf16)v;
            }
        }
        asm volatile("s_wait_dscnt 0" ::: "memory");
        __builtin_amdgcn_wave_barrier();

        // ---- GEMM 1: [16 x KPAD] x [KPAD x 64] + b1 (bias folded into C).
        v8f acc[4];
#pragma unroll
        for (int nt = 0; nt < 4; ++nt)
#pragma unroll
            for (int i = 0; i < 8; ++i) acc[nt][i] = b1n[nt];

#pragma unroll
        for (int kc = 0; kc < NKC1; ++kc) {
            const __bf16* ap = tileA + n0 * KPAD + kc * 32 + half * 16;
            v8bf lo = *reinterpret_cast<const v8bf*>(ap);
            v8bf hi = *reinterpret_cast<const v8bf*>(ap + 8);
            v16bf afrag = __builtin_shufflevector(lo, hi,
                0, 1, 2, 3, 4, 5, 6, 7, 8, 9, 10, 11, 12, 13, 14, 15);
#pragma unroll
            for (int nt = 0; nt < 4; ++nt)
                acc[nt] = __builtin_amdgcn_wmma_f32_16x16x32_bf16(
                    false, afrag, false, w1f[kc][nt], (short)0, acc[nt],
                    false, false);
        }

        // ---- relu (v_med3_f32) -> bf16 y tile in LDS ([16 x 64]).
#pragma unroll
        for (int nt = 0; nt < 4; ++nt)
#pragma unroll
            for (int i = 0; i < 8; ++i) {
                const float v = relu1(acc[nt][i]);
                const int m = half * 8 + i;           // C/D row mapping
                tileY[m * HID + nt * 16 + n0] = (__bf16)v;
            }
        asm volatile("s_wait_dscnt 0" ::: "memory");
        __builtin_amdgcn_wave_barrier();

        // ---- GEMM 2: [16 x 64] x [64 x 64] + b2.
        v8f acc2[4];
#pragma unroll
        for (int nt = 0; nt < 4; ++nt)
#pragma unroll
            for (int i = 0; i < 8; ++i) acc2[nt][i] = b2n[nt];

#pragma unroll
        for (int kc = 0; kc < 2; ++kc) {
            const __bf16* ap = tileY + n0 * HID + kc * 32 + half * 16;
            v8bf lo = *reinterpret_cast<const v8bf*>(ap);
            v8bf hi = *reinterpret_cast<const v8bf*>(ap + 8);
            v16bf afrag = __builtin_shufflevector(lo, hi,
                0, 1, 2, 3, 4, 5, 6, 7, 8, 9, 10, 11, 12, 13, 14, 15);
#pragma unroll
            for (int nt = 0; nt < 4; ++nt)
                acc2[nt] = __builtin_amdgcn_wmma_f32_16x16x32_bf16(
                    false, afrag, false, w2f[kc][nt], (short)0, acc2[nt],
                    false, false);
        }

        // ---- relu + segment-max scatter (uint atomicMax == float max, v>=0).
        if (full) {
#pragma unroll
            for (int i = 0; i < 8; ++i) {
                const int m = half * 8 + i;
                const long long d = dst_idx[ebase + m];
                unsigned int* arow =
                    reinterpret_cast<unsigned int*>(agg + (size_t)d * HID);
#pragma unroll
                for (int nt = 0; nt < 4; ++nt) {
                    const float v = relu1(acc2[nt][i]);
                    atomicMax(&arow[nt * 16 + n0], __float_as_uint(v));
                }
            }
        } else {
#pragma unroll
            for (int i = 0; i < 8; ++i) {
                const int m = half * 8 + i;
                const int e = ebase + m;
                if (e < n_edges) {
                    const long long d = dst_idx[e];
                    unsigned int* arow =
                        reinterpret_cast<unsigned int*>(agg + (size_t)d * HID);
#pragma unroll
                    for (int nt = 0; nt < 4; ++nt) {
                        const float v = relu1(acc2[nt][i]);
                        atomicMax(&arow[nt * 16 + n0], __float_as_uint(v));
                    }
                }
            }
        }
        __builtin_amdgcn_wave_barrier();
    }
}

// ---------------------------------------------------------------------------
// Head: out = h @ head_w[64x2] + head_b. Memory-bound, plain VALU.
// ---------------------------------------------------------------------------
__global__ void head_kernel(const float* __restrict__ h,
                            const float* __restrict__ w,
                            const float* __restrict__ b,
                            float* __restrict__ out, int n_nodes) {
    int i = blockIdx.x * blockDim.x + threadIdx.x;
    if (i >= n_nodes) return;
    float a0 = b[0], a1 = b[1];
    const float* hr = h + (size_t)i * HID;
#pragma unroll
    for (int k = 0; k < HID; ++k) {
        float hv = hr[k];
        a0 = fmaf(hv, w[k * 2 + 0], a0);
        a1 = fmaf(hv, w[k * 2 + 1], a1);
    }
    out[(size_t)i * 2 + 0] = a0;
    out[(size_t)i * 2 + 1] = a1;
}

// ---------------------------------------------------------------------------
extern "C" void kernel_launch(void* const* d_in, const int* in_sizes, int n_in,
                              void* d_out, int out_size, void* d_ws, size_t ws_size,
                              hipStream_t stream) {
    const float*     x     = (const float*)d_in[0];
    const float*     pos   = (const float*)d_in[1];
    const long long* eidx  = (const long long*)d_in[2];   // int64 [2, E]
    const float*     w1[3] = {(const float*)d_in[3],  (const float*)d_in[7],  (const float*)d_in[11]};
    const float*     b1[3] = {(const float*)d_in[4],  (const float*)d_in[8],  (const float*)d_in[12]};
    const float*     w2[3] = {(const float*)d_in[5],  (const float*)d_in[9],  (const float*)d_in[13]};
    const float*     b2[3] = {(const float*)d_in[6],  (const float*)d_in[10], (const float*)d_in[14]};
    const float*     head_w = (const float*)d_in[15];
    const float*     head_b = (const float*)d_in[16];
    float*           out    = (float*)d_out;

    const int n_nodes = in_sizes[0] / 8;     // IN_DIM = 8
    const int n_edges = in_sizes[2] / 2;
    const long long* src = eidx;
    const long long* dst = eidx + n_edges;

    // Workspace layout.
    char* ws = (char*)d_ws;
    const size_t hbytes = (size_t)n_nodes * HID * sizeof(float);
    size_t off = 0;
    float* bufA = (float*)(ws + off); off += (hbytes + 255) & ~(size_t)255;
    float* bufB = (float*)(ws + off); off += (hbytes + 255) & ~(size_t)255;
    unsigned int* w1p[3];
    unsigned int* w2p[3];
    const size_t w1p_bytes = (size_t)3 * 4 * 32 * 8 * 4;   // up to 3 K-chunks
    const size_t w2p_bytes = (size_t)2 * 4 * 32 * 8 * 4;
    for (int l = 0; l < 3; ++l) {
        w1p[l] = (unsigned int*)(ws + off); off += (w1p_bytes + 255) & ~(size_t)255;
        w2p[l] = (unsigned int*)(ws + off); off += (w2p_bytes + 255) & ~(size_t)255;
    }
    (void)ws_size;

    // Pack weights into WMMA B-fragment layout.
    const int d_ins[3] = {8, HID, HID};
    for (int l = 0; l < 3; ++l) {
        const int Kact = d_ins[l] + 3;
        const int nkc1 = (Kact + 31) / 32;
        {
            int threads = nkc1 * 4 * 32;
            pack_b_kernel<<<(threads + 127) / 128, 128, 0, stream>>>(w1[l], w1p[l], Kact, nkc1);
        }
        {
            int threads = 2 * 4 * 32;
            pack_b_kernel<<<(threads + 127) / 128, 128, 0, stream>>>(w2[l], w2p[l], HID, 2);
        }
    }

    const size_t hcount4 = (size_t)n_nodes * HID / 4;
    const int layer_blocks = 2000;    // 16K waves over 100K edge tiles

    // Layer 0: x -> bufA
    zero_kernel<<<1024, 256, 0, stream>>>((float4*)bufA, hcount4);
    pointnet_layer_kernel<8, 32, 1><<<layer_blocks, 256, 0, stream>>>(
        x, pos, src, dst, w1p[0], b1[0], w2p[0], b2[0], bufA, n_edges);

    // Layer 1: bufA -> bufB
    zero_kernel<<<1024, 256, 0, stream>>>((float4*)bufB, hcount4);
    pointnet_layer_kernel<HID, 96, 3><<<layer_blocks, 256, 0, stream>>>(
        bufA, pos, src, dst, w1p[1], b1[1], w2p[1], b2[1], bufB, n_edges);

    // Layer 2: bufB -> bufA
    zero_kernel<<<1024, 256, 0, stream>>>((float4*)bufA, hcount4);
    pointnet_layer_kernel<HID, 96, 3><<<layer_blocks, 256, 0, stream>>>(
        bufB, pos, src, dst, w1p[2], b1[2], w2p[2], b2[2], bufA, n_edges);

    // Head: bufA -> out
    head_kernel<<<(n_nodes + 255) / 256, 256, 0, stream>>>(
        bufA, head_w, head_b, out, n_nodes);
}
